// Layer_51101520888167
// MI455X (gfx1250) — matrix operand order repaired and checked by reading, not compiled, not run on hardware
//
#include <hip/hip_runtime.h>
#include <math.h>

// ---------------------------------------------------------------------------
// B=16,T=2048 -> M=32768 rows
//   GEMM1: M x 512 @ 512 x 640 ; softmax over V=320 per (row,g) ; GEMM2: M x 1024 @ 1024 x 512
// ---------------------------------------------------------------------------
#define MROWS 32768
#define DIN   512
#define N1    640
#define NG    2
#define NV    320
#define DW    512
#define K2    1024

typedef __bf16 bf16_t;
typedef __attribute__((ext_vector_type(16))) __bf16        v16bf;
typedef __attribute__((ext_vector_type(8)))  __bf16        bf16x8;
typedef __attribute__((ext_vector_type(8)))  float         v8f;
typedef __attribute__((ext_vector_type(4)))  float         f32x4;
typedef __attribute__((ext_vector_type(4)))  unsigned int  u32x4;

union Frag {            // 32 bytes: one WMMA A/B operand per lane
    v16bf v;
    u32x4 q[2];
};

__device__ __forceinline__ void splitbf(float x, bf16_t& hi, bf16_t& lo) {
    hi = (bf16_t)x;
    lo = (bf16_t)(x - (float)hi);
}

// One 16x16x32 split-f32 step: acc += A*B with A,B given as (hi,lo) bf16 pairs.
__device__ __forceinline__ v8f wmma_split(v16bf ahi, v16bf alo, v16bf bhi, v16bf blo, v8f acc) {
    acc = __builtin_amdgcn_wmma_f32_16x16x32_bf16(false, ahi, false, bhi, (short)0, acc, false, false);
    acc = __builtin_amdgcn_wmma_f32_16x16x32_bf16(false, ahi, false, blo, (short)0, acc, false, false);
    acc = __builtin_amdgcn_wmma_f32_16x16x32_bf16(false, alo, false, bhi, (short)0, acc, false, false);
    return acc;
}

// Convert 8 consecutive f32 (two f32x4) into hi/lo bf16x8 and store to LDS (16B each).
__device__ __forceinline__ void stage8(const f32x4 a, const f32x4 b,
                                       bf16_t* __restrict__ hDst, bf16_t* __restrict__ lDst) {
    bf16x8 h, l;
    #pragma unroll
    for (int e = 0; e < 4; ++e) { bf16_t hh, ll; splitbf(a[e], hh, ll); h[e]   = hh; l[e]   = ll; }
    #pragma unroll
    for (int e = 0; e < 4; ++e) { bf16_t hh, ll; splitbf(b[e], hh, ll); h[4+e] = hh; l[4+e] = ll; }
    *(bf16x8*)hDst = h;
    *(bf16x8*)lDst = l;
}

// A fragment: rows in LDS as [row][32]; lane layout per ISA (lanes<16: K 0-7/16-23; lanes>=16: +8)
__device__ __forceinline__ v16bf loadAfrag(const bf16_t* __restrict__ arr, int lr, int kbA) {
    Frag f;
    const bf16_t* base = arr + lr * 32 + kbA;
    f.q[0] = *(const u32x4*)(base);        // K = kbA .. kbA+7
    f.q[1] = *(const u32x4*)(base + 16);   // K = kbA+16 .. kbA+23
    return f.v;
}

// B fragment: LDS stored N-major [n][32]; lane layout per ISA (lanes<16: K 0-7/16-23; lanes>=16: +8)
__device__ __forceinline__ v16bf loadBfrag(const bf16_t* __restrict__ arr, int n, int kbB) {
    Frag f;
    const bf16_t* base = arr + n * 32 + kbB;
    f.q[0] = *(const u32x4*)(base);
    f.q[1] = *(const u32x4*)(base + 16);
    return f.v;
}

// ---------------------------------------------------------------------------
// Kernel 1: h = relu(data @ W1 + b1), stored permuted as logits (B,T,G,V).
//   grid = (10, 128): x -> 64-col group, y -> 256-row group. 256 threads = 8 waves.
//   Each wave: 32 rows (2 M-tiles) x 64 cols (4 N-tiles); K-loop: 16 steps of K=32.
// ---------------------------------------------------------------------------
__global__ __launch_bounds__(256) void gemm1_logits_kernel(
    const float* __restrict__ data, const float* __restrict__ W1,
    const float* __restrict__ b1, float* __restrict__ outP)
{
    __shared__ __align__(16) bf16_t Ahi[256 * 32];   // [row][k], 16KB
    __shared__ __align__(16) bf16_t Alo[256 * 32];
    __shared__ __align__(16) bf16_t Bhi[64 * 32];    // [n][k] (N-major), 4KB
    __shared__ __align__(16) bf16_t Blo[64 * 32];

    const int t     = threadIdx.x;
    const int wave  = t >> 5;
    const int lane  = t & 31;
    const int n0    = blockIdx.x * 64;
    const int mBase = blockIdx.y * 256;

    v8f acc0[4] = {};
    v8f acc1[4] = {};

    const int nlS  = t & 63;    // B staging: column 0..63
    const int ksS  = t >> 6;    // B staging: 8-elem k segment 0..3

    for (int k0 = 0; k0 < DIN; k0 += 32) {
        // ---- stage A: one full 32-elem k-slab of one row per thread ----
        {
            const f32x4* src = (const f32x4*)(data + (size_t)(mBase + t) * DIN + k0);
            if (k0 + 32 < DIN) __builtin_prefetch((const float*)src + 32, 0, 1);
            #pragma unroll
            for (int j = 0; j < 4; ++j)
                stage8(src[2*j], src[2*j+1], &Ahi[t*32 + j*8], &Alo[t*32 + j*8]);
        }
        // ---- stage B (transposed into [n][k]): coalesced global reads per i ----
        {
            bf16x8 h, l;
            #pragma unroll
            for (int i = 0; i < 8; ++i) {
                const float v = W1[(size_t)(k0 + ksS*8 + i) * N1 + n0 + nlS];
                bf16_t hh, ll; splitbf(v, hh, ll); h[i] = hh; l[i] = ll;
            }
            *(bf16x8*)&Bhi[nlS*32 + ksS*8] = h;
            *(bf16x8*)&Blo[nlS*32 + ksS*8] = l;
        }
        __syncthreads();

        const int lr0 = wave * 32 + (lane & 15);
        const int lr1 = lr0 + 16;
        const int kbA = (lane < 16) ? 0 : 8;
        const int kbB = (lane < 16) ? 0 : 8;
        const int nl  = lane & 15;

        const v16bf a0h = loadAfrag(Ahi, lr0, kbA);
        const v16bf a0l = loadAfrag(Alo, lr0, kbA);
        const v16bf a1h = loadAfrag(Ahi, lr1, kbA);
        const v16bf a1l = loadAfrag(Alo, lr1, kbA);

        #pragma unroll
        for (int nt = 0; nt < 4; ++nt) {
            const v16bf bh = loadBfrag(Bhi, nt*16 + nl, kbB);
            const v16bf bl = loadBfrag(Blo, nt*16 + nl, kbB);
            acc0[nt] = wmma_split(a0h, a0l, bh, bl, acc0[nt]);
            acc1[nt] = wmma_split(a1h, a1l, bh, bl, acc1[nt]);
        }
        __syncthreads();
    }

    // ---- epilogue: bias + relu, permuted store: col n -> (g = n&1, v = n>>1) ----
    const int row0 = mBase + wave * 32 + ((lane < 16) ? 0 : 8);
    const int nlan = lane & 15;
    #pragma unroll
    for (int nt = 0; nt < 4; ++nt) {
        const int n    = n0 + nt * 16 + nlan;
        const float bs = b1[n];
        const int g = n & 1, v = n >> 1;
        #pragma unroll
        for (int r = 0; r < 8; ++r) {
            float v0 = acc0[nt][r] + bs;
            float v1 = acc1[nt][r] + bs;
            v0 = v0 > 0.f ? v0 : 0.f;
            v1 = v1 > 0.f ? v1 : 0.f;
            outP[(size_t)(row0 + r)      * N1 + g * NV + v] = v0;
            outP[(size_t)(row0 + 16 + r) * N1 + g * NV + v] = v1;
        }
    }
}

// ---------------------------------------------------------------------------
// Kernel 2: per (b,t,g) row r: p = softmax(logits + gumbel)/2 in place over the
//   logits region of d_out; argmax index + straight-through weight into ws.
//   One wave32 per row (V=320 -> 10 elems/lane), shuffle reductions.
// ---------------------------------------------------------------------------
__global__ __launch_bounds__(256) void gumbel_softmax_kernel(
    float* __restrict__ P, const float* __restrict__ gum,
    int* __restrict__ Idx, float* __restrict__ Wt)
{
    const int lane = threadIdx.x & 31;
    const int r    = blockIdx.x * 8 + (threadIdx.x >> 5);
    const size_t base = (size_t)r * NV;   // logits flat layout == gumbel flat layout

    float x[10];
    float xmax = -3.4e38f;
    #pragma unroll
    for (int j = 0; j < 10; ++j) {
        const int v = lane + 32 * j;
        x[j] = P[base + v] + gum[base + v];
        xmax = fmaxf(xmax, x[j]);
    }
    #pragma unroll
    for (int off = 16; off > 0; off >>= 1)
        xmax = fmaxf(xmax, __shfl_xor(xmax, off, 32));

    float s = 0.f;
    #pragma unroll
    for (int j = 0; j < 10; ++j) { x[j] = expf(x[j] - xmax); s += x[j]; }
    #pragma unroll
    for (int off = 16; off > 0; off >>= 1)
        s += __shfl_xor(s, off, 32);

    const float inv = 1.0f / (s * 2.0f);   // softmax / tau

    float pm = -1.f; int im = NV;
    #pragma unroll
    for (int j = 0; j < 10; ++j) {
        const int v   = lane + 32 * j;
        const float p = x[j] * inv;
        P[base + v] = p;
        if (p > pm || (p == pm && v < im)) { pm = p; im = v; }
    }
    #pragma unroll
    for (int off = 16; off > 0; off >>= 1) {
        const float ov = __shfl_xor(pm, off, 32);
        const int   oi = __shfl_xor(im, off, 32);
        if (ov > pm || (ov == pm && oi < im)) { pm = ov; im = oi; }
    }
    if (lane == 0) {
        Idx[r] = im;
        Wt[r]  = (1.0f - pm) + pm;   // straight-through forward value at the argmax
    }
}

// ---------------------------------------------------------------------------
// Kernel 3: q = relu(sub_words @ W2 + b2); sub_words rows gathered on the fly:
//   row m, k-slab k0: g = k0>>9, A slice = Wt[m,g] * code_book[g, Idx[m,g], k0&511 ..]
//   grid = (8, 128): x -> 64-col group, y -> 256-row group.
// ---------------------------------------------------------------------------
__global__ __launch_bounds__(256) void gemm2_codebook_kernel(
    const float* __restrict__ cb, const float* __restrict__ W2,
    const float* __restrict__ b2, const int* __restrict__ Idx,
    const float* __restrict__ Wt, float* __restrict__ outQ)
{
    __shared__ __align__(16) bf16_t Ahi[256 * 32];
    __shared__ __align__(16) bf16_t Alo[256 * 32];
    __shared__ __align__(16) bf16_t Bhi[64 * 32];    // [n][k] N-major
    __shared__ __align__(16) bf16_t Blo[64 * 32];

    const int t     = threadIdx.x;
    const int wave  = t >> 5;
    const int lane  = t & 31;
    const int n0    = blockIdx.x * 64;
    const int mBase = blockIdx.y * 256;

    v8f acc0[4] = {};
    v8f acc1[4] = {};

    const int nlS = t & 63;
    const int ksS = t >> 6;

    for (int k0 = 0; k0 < K2; k0 += 32) {
        const int g  = k0 >> 9;
        const int d0 = k0 & 511;
        // ---- stage A: weighted codebook row slab, one row per thread ----
        {
            const int   m   = mBase + t;
            const int   idx = Idx[m * NG + g];
            const float w   = Wt[m * NG + g];
            const f32x4* src = (const f32x4*)(cb + ((size_t)g * NV + idx) * DW + d0);
            #pragma unroll
            for (int j = 0; j < 4; ++j) {
                f32x4 a = src[2*j] * w;
                f32x4 b = src[2*j+1] * w;
                stage8(a, b, &Ahi[t*32 + j*8], &Alo[t*32 + j*8]);
            }
        }
        // ---- stage B (W2, transposed into [n][k]) ----
        {
            bf16x8 h, l;
            #pragma unroll
            for (int i = 0; i < 8; ++i) {
                const float v = W2[(size_t)(k0 + ksS*8 + i) * DW + n0 + nlS];
                bf16_t hh, ll; splitbf(v, hh, ll); h[i] = hh; l[i] = ll;
            }
            *(bf16x8*)&Bhi[nlS*32 + ksS*8] = h;
            *(bf16x8*)&Blo[nlS*32 + ksS*8] = l;
        }
        __syncthreads();

        const int lr0 = wave * 32 + (lane & 15);
        const int lr1 = lr0 + 16;
        const int kbA = (lane < 16) ? 0 : 8;
        const int kbB = (lane < 16) ? 0 : 8;
        const int nl  = lane & 15;

        const v16bf a0h = loadAfrag(Ahi, lr0, kbA);
        const v16bf a0l = loadAfrag(Alo, lr0, kbA);
        const v16bf a1h = loadAfrag(Ahi, lr1, kbA);
        const v16bf a1l = loadAfrag(Alo, lr1, kbA);

        #pragma unroll
        for (int nt = 0; nt < 4; ++nt) {
            const v16bf bh = loadBfrag(Bhi, nt*16 + nl, kbB);
            const v16bf bl = loadBfrag(Blo, nt*16 + nl, kbB);
            acc0[nt] = wmma_split(a0h, a0l, bh, bl, acc0[nt]);
            acc1[nt] = wmma_split(a1h, a1l, bh, bl, acc1[nt]);
        }
        __syncthreads();
    }

    const int row0 = mBase + wave * 32 + ((lane < 16) ? 0 : 8);
    const int nlan = lane & 15;
    #pragma unroll
    for (int nt = 0; nt < 4; ++nt) {
        const int n    = n0 + nt * 16 + nlan;
        const float bs = b2[n];
        #pragma unroll
        for (int r = 0; r < 8; ++r) {
            float v0 = acc0[nt][r] + bs;
            float v1 = acc1[nt][r] + bs;
            v0 = v0 > 0.f ? v0 : 0.f;
            v1 = v1 > 0.f ? v1 : 0.f;
            outQ[(size_t)(row0 + r)      * DW + n] = v0;
            outQ[(size_t)(row0 + 16 + r) * DW + n] = v1;
        }
    }
}

// ---------------------------------------------------------------------------
extern "C" void kernel_launch(void* const* d_in, const int* in_sizes, int n_in,
                              void* d_out, int out_size, void* d_ws, size_t ws_size,
                              hipStream_t stream) {
    const float* data = (const float*)d_in[0];   // (16,2048,512)
    const float* gum  = (const float*)d_in[1];   // (16,2048,2,320)
    const float* W1   = (const float*)d_in[2];   // (512,640)
    const float* b1   = (const float*)d_in[3];   // (640,)
    const float* W2   = (const float*)d_in[4];   // (1024,512)
    const float* b2   = (const float*)d_in[5];   // (512,)
    const float* cb   = (const float*)d_in[6];   // (2,320,512)

    float* out  = (float*)d_out;
    float* outP = out;                              // p_g_v: 32768*640 floats
    float* outQ = out + (size_t)MROWS * N1;         // q:     32768*512 floats

    int*   Idx = (int*)d_ws;                                    // 65536 ints
    float* Wt  = (float*)((char*)d_ws + (size_t)MROWS * NG * sizeof(int));

    gemm1_logits_kernel<<<dim3(N1 / 64, MROWS / 256), 256, 0, stream>>>(data, W1, b1, outP);
    gumbel_softmax_kernel<<<dim3((MROWS * NG) / 8), 256, 0, stream>>>(outP, gum, Idx, Wt);
    gemm2_codebook_kernel<<<dim3(DW / 64, MROWS / 256), 256, 0, stream>>>(cb, W2, b2, Idx, Wt, outQ);
}
